// CNNConcatLinear_53240414601643
// MI455X (gfx1250) — matrix operand order repaired and checked by reading, not compiled, not run on hardware
//
#include <hip/hip_runtime.h>
#include <cstdint>
#include <cstddef>

// ---------------------------------------------------------------------------
// Types / helpers
// ---------------------------------------------------------------------------
typedef __attribute__((ext_vector_type(16))) __bf16 v16bf;
typedef __attribute__((ext_vector_type(8)))  float  v8f;
typedef unsigned short ushort_t;

struct __align__(16) U4 { unsigned int a, b, c, d; };

union FragBF {
    v16bf    v;
    ushort_t u[16];
    U4       q[2];
};

__device__ __forceinline__ ushort_t f2bf(float f) {
    unsigned u = __float_as_uint(f);
    unsigned r = u + 0x7FFFu + ((u >> 16) & 1u);   // round-to-nearest-even
    return (ushort_t)(r >> 16);
}
__device__ __forceinline__ float bf2f(ushort_t h) {
    return __uint_as_float(((unsigned)h) << 16);
}

#define WMMA_BF16(A, B, C) \
    __builtin_amdgcn_wmma_f32_16x16x32_bf16(false, (A), false, (B), (short)0, (C), false, false)

// Problem constants
#define BB  1024
#define NN  24
#define CC  512
#define D2C 1024   // 2*C

// LDS row strides (elements), padded to break bank-group aliasing while
// keeping every row start 16B-aligned.
#define HS  1032   // h rows    (1024 cols)
#define TS  1032   // trans rows(1024 cols)
#define T3S 520    // t3 rows   (512 cols)
#define T4S 264    // t4 rows   (256 cols)

// ---------------------------------------------------------------------------
// Kernel 0a: elementwise f32 -> bf16 weight conversion
// ---------------------------------------------------------------------------
__global__ void __launch_bounds__(256) cvt_bf16_kernel(
    const float* __restrict__ src, ushort_t* __restrict__ dst, int n)
{
    int i = blockIdx.x * 256 + threadIdx.x;
    if (i < n) dst[i] = f2bf(src[i]);
}

// ---------------------------------------------------------------------------
// Kernel 0b: conv weight [O][C][K] f32 -> [O][K][C] bf16 (reduction-contiguous)
// ---------------------------------------------------------------------------
__global__ void __launch_bounds__(256) cvt_tr_conv_kernel(
    const float* __restrict__ src, ushort_t* __restrict__ dst, int Cout, int K)
{
    int total = Cout * K * D2C;
    for (int i = blockIdx.x * 256 + threadIdx.x; i < total; i += gridDim.x * 256) {
        int c = i & (D2C - 1);
        int t = i >> 10;
        int k = t % K;
        int o = t / K;
        dst[i] = f2bf(src[((size_t)o * D2C + c) * K + k]);
    }
}

// ---------------------------------------------------------------------------
// Kernel 1: new_ctx (attention-over-3 context pooling) -> bf16 [B,512]
// ---------------------------------------------------------------------------
__global__ void __launch_bounds__(256) newctx_kernel(
    const float* __restrict__ beta, const float* __restrict__ ctx,
    const float* __restrict__ emb_w, const float* __restrict__ emb_b,
    const float* __restrict__ demb_w, const float* __restrict__ demb_b,
    ushort_t* __restrict__ ncbf)
{
    int b = blockIdx.x;
    // Fold conv1d(1x1, 3->64) then conv1d(1x1, 64->3) into 3x3 matrix + bias.
    float M[3][3], bv[3];
    #pragma unroll
    for (int i = 0; i < 3; ++i) {
        float s0 = 0.f, s1 = 0.f, s2 = 0.f, sb = demb_b[i];
        for (int j = 0; j < 64; ++j) {
            float dw = demb_w[i * 64 + j];
            s0 += dw * emb_w[j * 3 + 0];
            s1 += dw * emb_w[j * 3 + 1];
            s2 += dw * emb_w[j * 3 + 2];
            sb += dw * emb_b[j];
        }
        M[i][0] = s0; M[i][1] = s1; M[i][2] = s2; bv[i] = sb;
    }
    float be = beta[b];
    float t0 = be, t1 = __sinf(be), t2 = __cosf(be);

    for (int c = threadIdx.x; c < CC; c += 256) {
        float cv = ctx[b * CC + c];
        float e0 = cv + t0, e1 = cv + t1, e2 = cv + t2;
        float a0 = M[0][0]*e0 + M[0][1]*e1 + M[0][2]*e2 + bv[0];
        float a1 = M[1][0]*e0 + M[1][1]*e1 + M[1][2]*e2 + bv[1];
        float a2 = M[2][0]*e0 + M[2][1]*e1 + M[2][2]*e2 + bv[2];
        float mx = fmaxf(a0, fmaxf(a1, a2));
        float x0 = __expf(a0 - mx), x1 = __expf(a1 - mx), x2 = __expf(a2 - mx);
        float inv = 1.f / (x0 + x1 + x2);
        float nc = (x0 * e0 + x1 * e1 + x2 * e2) * inv;
        ncbf[b * CC + c] = f2bf(nc);
    }
}

// ---------------------------------------------------------------------------
// Kernel 2: batched gate projection   out[B,Dout] = f(ncbf[B,512] @ W^T + gb)
// W is bf16 [Dout,512]. One 16x16 tile per wave; K-loop = 16 WMMAs; the
// B operand loads are unconditional from a clamped row (garbage B columns
// only pollute D columns that are never stored).
// ---------------------------------------------------------------------------
__global__ void __launch_bounds__(256) gates_kernel(
    const ushort_t* __restrict__ A, const ushort_t* __restrict__ W,
    const float* __restrict__ gb, int Dout, int dosig, float* __restrict__ out)
{
    int wave = threadIdx.x >> 5, lane = threadIdx.x & 31;
    int half = lane >> 4, nIdx = lane & 15;
    int mt = blockIdx.y * 8 + wave;      // 64 M-tiles cover B=1024
    int nt = blockIdx.x;
    int o  = nt * 16 + nIdx;
    int m  = mt * 16 + nIdx;
    bool ov = (o < Dout);
    const ushort_t* wrow = W + (size_t)(ov ? o : 0) * CC;

    v8f acc = {0.f,0.f,0.f,0.f,0.f,0.f,0.f,0.f};
    for (int w32 = 0; w32 < CC / 32; ++w32) {
        FragBF fa, fb;
        int abase = m * CC + w32 * 32 + half * 8;
        fa.q[0] = *(const U4*)(A + abase);
        fa.q[1] = *(const U4*)(A + abase + 16);
        int bbase = w32 * 32 + half * 8;
        fb.q[0] = *(const U4*)(wrow + bbase);
        fb.q[1] = *(const U4*)(wrow + bbase + 16);
        acc = WMMA_BF16(fa.v, fb.v, acc);
    }
    if (ov) {
        float bias = gb ? gb[o] : 0.f;
        #pragma unroll
        for (int v = 0; v < 8; ++v) {
            float val = acc[v] + bias;
            if (dosig) val = 1.f / (1.f + __expf(-val));
            int mm = mt * 16 + half * 8 + v;
            out[(size_t)mm * Dout + o] = val;
        }
    }
}

// ---------------------------------------------------------------------------
// Kernel 3: per-batch fused pipeline (h -> 6 convs -> CSL3 -> CSL4 -> CSLl)
// 8 waves, ~162KB dynamic LDS, all GEMMs via v_wmma_f32_16x16x32_bf16.
// ---------------------------------------------------------------------------
__global__ void __launch_bounds__(256) main_kernel(
    const float* __restrict__ x,
    const float* __restrict__ c1_w, const float* __restrict__ c1_b,
    const float* __restrict__ G1,   const float* __restrict__ H1,
    const ushort_t* __restrict__ c3wb, const float* __restrict__ c3_b,
    const float* __restrict__ G3,   const float* __restrict__ H3,
    const ushort_t* __restrict__ c4wb, const float* __restrict__ c4_b,
    const float* __restrict__ G4,   const float* __restrict__ H4,
    const float* __restrict__ cl_w, const float* __restrict__ cl_b,
    const float* __restrict__ GL,   const float* __restrict__ HL,
    const ushort_t* __restrict__ cw1, const float* __restrict__ b1,
    const ushort_t* __restrict__ cw2, const float* __restrict__ b2,
    const ushort_t* __restrict__ cw3, const float* __restrict__ b3,
    const ushort_t* __restrict__ cw4, const float* __restrict__ b4,
    const ushort_t* __restrict__ cw5, const float* __restrict__ b5,
    const ushort_t* __restrict__ cw6, const float* __restrict__ b6,
    float* __restrict__ out)
{
    extern __shared__ ushort_t smem[];
    ushort_t* sm_h     = smem;                    // 24 x HS
    ushort_t* sm_trans = sm_h     + NN * HS;      // 32 x TS
    ushort_t* sm_t3    = sm_trans + 32 * TS;      // 32 x T3S
    ushort_t* sm_t4    = sm_t3    + 32 * T3S;     // 32 x T4S

    int b = blockIdx.x, tid = threadIdx.x;
    int wave = tid >> 5, lane = tid & 31, half = lane >> 4, nIdx = lane & 15;
    const U4 Z4 = {0u, 0u, 0u, 0u};

    // ---- Phase 1: h = (x @ c1_w^T + c1_b) * G1 + H1 + PE  -> LDS bf16 ----
    const float NEG_LOG1E4_OVER_D = -9.210340371976184f / 1024.f;
    for (int idx = tid; idx < NN * D2C; idx += 256) {
        int n = idx >> 10, c = idx & 1023;
        float x0 = x[b * 48 + n * 2], x1 = x[b * 48 + n * 2 + 1];
        float lin = x0 * c1_w[c * 2] + x1 * c1_w[c * 2 + 1] + c1_b[c];
        float val = lin * G1[b * D2C + c] + H1[b * D2C + c];
        float freq = __expf((float)(c & ~1) * NEG_LOG1E4_OVER_D);
        float ang = (float)n * freq;
        val += (c & 1) ? __cosf(ang) : __sinf(ang);
        sm_h[n * HS + c] = f2bf(val);
    }
    __syncthreads();

    // ---- Phase 2: six convs as implicit-im2col WMMA GEMMs -> sm_trans ----
    // 128 jobs = 2 M-tiles x 64 output-channel tiles, round-robin over waves.
    for (int job = wave; job < 128; job += 8) {
        int mt = job & 1, ct = job >> 1;
        const ushort_t* wj; const float* bj; int Kj, tbase;
        if      (ct < 32) { tbase = 0;  wj = cw1; bj = b1; Kj = 1;  }
        else if (ct < 48) { tbase = 32; wj = cw2; bj = b2; Kj = 3;  }
        else if (ct < 52) { tbase = 48; wj = cw3; bj = b3; Kj = 5;  }
        else if (ct < 56) { tbase = 52; wj = cw4; bj = b4; Kj = 7;  }
        else if (ct < 60) { tbase = 56; wj = cw5; bj = b5; Kj = 9;  }
        else              { tbase = 60; wj = cw6; bj = b6; Kj = 11; }
        int pad = Kj >> 1;
        int olocal = (ct - tbase) * 16 + nIdx;

        v8f acc = {0.f,0.f,0.f,0.f,0.f,0.f,0.f,0.f};
        for (int k = 0; k < Kj; ++k) {
            int row = mt * 16 + nIdx + k - pad;                // h row (zero-padded)
            bool inr = (row >= 0) && (row < NN);
            size_t wb = ((size_t)olocal * Kj + k) << 10;       // [o][k][c] bf16
            for (int w32 = 0; w32 < D2C / 32; ++w32) {
                FragBF fa, fb;
                if (inr) {
                    int abase = row * HS + w32 * 32 + half * 8;
                    fa.q[0] = *(const U4*)(sm_h + abase);
                    fa.q[1] = *(const U4*)(sm_h + abase + 16);
                } else {
                    fa.q[0] = Z4; fa.q[1] = Z4;
                }
                size_t bofs = wb + w32 * 32 + half * 8;
                fb.q[0] = *(const U4*)(wj + bofs);
                fb.q[1] = *(const U4*)(wj + bofs + 16);
                __builtin_prefetch(wj + bofs + 64, 0, 0);
                acc = WMMA_BF16(fa.v, fb.v, acc);              // EXEC all-ones here
            }
        }
        float bias = bj[olocal];
        int ocol = ct * 16 + nIdx;
        #pragma unroll
        for (int v = 0; v < 8; ++v) {
            int m = mt * 16 + half * 8 + v;                    // rows >=24 unused later
            sm_trans[m * TS + ocol] = f2bf(acc[v] + bias);
        }
    }
    __syncthreads();

    // ---- Phase 3: CSL3  t3 = (trans @ c3_w^T + c3_b)*G3 + H3  -> sm_t3 ----
    for (int job = wave; job < 64; job += 8) {
        int mt = job >> 5, nt = job & 31;
        int o = nt * 16 + nIdx, arow = mt * 16 + nIdx;
        const ushort_t* wrow = c3wb + (size_t)o * D2C;
        v8f acc = {0.f,0.f,0.f,0.f,0.f,0.f,0.f,0.f};
        for (int w32 = 0; w32 < D2C / 32; ++w32) {
            FragBF fa, fb;
            int abase = arow * TS + w32 * 32 + half * 8;
            fa.q[0] = *(const U4*)(sm_trans + abase);
            fa.q[1] = *(const U4*)(sm_trans + abase + 16);
            int bbase = w32 * 32 + half * 8;
            fb.q[0] = *(const U4*)(wrow + bbase);
            fb.q[1] = *(const U4*)(wrow + bbase + 16);
            acc = WMMA_BF16(fa.v, fb.v, acc);
        }
        float g = G3[b * CC + o], hb = H3[b * CC + o], cb = c3_b[o];
        #pragma unroll
        for (int v = 0; v < 8; ++v) {
            int m = mt * 16 + half * 8 + v;
            sm_t3[m * T3S + o] = f2bf((acc[v] + cb) * g + hb);
        }
    }
    __syncthreads();

    // ---- Phase 4: CSL4  t4 = (t3 @ c4_w^T + c4_b)*G4 + H4  -> sm_t4 ----
    for (int job = wave; job < 32; job += 8) {
        int mt = job >> 4, nt = job & 15;
        int o = nt * 16 + nIdx, arow = mt * 16 + nIdx;
        const ushort_t* wrow = c4wb + (size_t)o * CC;
        v8f acc = {0.f,0.f,0.f,0.f,0.f,0.f,0.f,0.f};
        for (int w32 = 0; w32 < CC / 32; ++w32) {
            FragBF fa, fb;
            int abase = arow * T3S + w32 * 32 + half * 8;
            fa.q[0] = *(const U4*)(sm_t3 + abase);
            fa.q[1] = *(const U4*)(sm_t3 + abase + 16);
            int bbase = w32 * 32 + half * 8;
            fb.q[0] = *(const U4*)(wrow + bbase);
            fb.q[1] = *(const U4*)(wrow + bbase + 16);
            acc = WMMA_BF16(fa.v, fb.v, acc);
        }
        float g = G4[b * 256 + o], hb = H4[b * 256 + o], cb = c4_b[o];
        #pragma unroll
        for (int v = 0; v < 8; ++v) {
            int m = mt * 16 + half * 8 + v;
            sm_t4[m * T4S + o] = f2bf((acc[v] + cb) * g + hb);
        }
    }
    __syncthreads();

    // ---- Phase 5: final CSL (256 -> 2), Dout too small for WMMA: VALU ----
    if (tid < NN * 2) {
        int n = tid >> 1, p = tid & 1;
        float s = 0.f;
        for (int k = 0; k < 256; ++k)
            s += bf2f(sm_t4[n * T4S + k]) * cl_w[p * 256 + k];
        out[b * 48 + n * 2 + p] =
            (s + cl_b[p]) * GL[b * 2 + p] + HL[b * 2 + p];
    }
}

// ---------------------------------------------------------------------------
// Host launcher
// ---------------------------------------------------------------------------
extern "C" void kernel_launch(void* const* d_in, const int* in_sizes, int n_in,
                              void* d_out, int out_size, void* d_ws, size_t ws_size,
                              hipStream_t stream) {
    (void)in_sizes; (void)n_in; (void)out_size; (void)ws_size;
    const float* x       = (const float*)d_in[0];
    const float* beta    = (const float*)d_in[1];
    const float* context = (const float*)d_in[2];
    const float* emb_w   = (const float*)d_in[3];
    const float* emb_b   = (const float*)d_in[4];
    const float* demb_w  = (const float*)d_in[5];
    const float* demb_b  = (const float*)d_in[6];
    const float* c1_w  = (const float*)d_in[7];
    const float* c1_b  = (const float*)d_in[8];
    const float* c1_gw = (const float*)d_in[9];
    const float* c1_gb = (const float*)d_in[10];
    const float* c1_hw = (const float*)d_in[11];
    const float* c3_w  = (const float*)d_in[12];
    const float* c3_b  = (const float*)d_in[13];
    const float* c3_gw = (const float*)d_in[14];
    const float* c3_gb = (const float*)d_in[15];
    const float* c3_hw = (const float*)d_in[16];
    const float* c4_w  = (const float*)d_in[17];
    const float* c4_b  = (const float*)d_in[18];
    const float* c4_gw = (const float*)d_in[19];
    const float* c4_gb = (const float*)d_in[20];
    const float* c4_hw = (const float*)d_in[21];
    const float* cl_w  = (const float*)d_in[22];
    const float* cl_b  = (const float*)d_in[23];
    const float* cl_gw = (const float*)d_in[24];
    const float* cl_gb = (const float*)d_in[25];
    const float* cl_hw = (const float*)d_in[26];
    const float* w1 = (const float*)d_in[27]; const float* b1 = (const float*)d_in[28];
    const float* w2 = (const float*)d_in[29]; const float* b2 = (const float*)d_in[30];
    const float* w3 = (const float*)d_in[31]; const float* b3 = (const float*)d_in[32];
    const float* w4 = (const float*)d_in[33]; const float* b4 = (const float*)d_in[34];
    const float* w5 = (const float*)d_in[35]; const float* b5 = (const float*)d_in[36];
    const float* w6 = (const float*)d_in[37]; const float* b6 = (const float*)d_in[38];
    float* out = (float*)d_out;

    // ---- Workspace layout ----
    char*     ws   = (char*)d_ws;
    ushort_t* ncbf = (ushort_t*)ws;                         // B*512 bf16 = 1 MB
    float*    G1   = (float*)(ws + (size_t)(1 << 20));
    float*    H1   = G1 + (size_t)BB * D2C;
    float*    G3   = H1 + (size_t)BB * D2C;
    float*    H3   = G3 + (size_t)BB * CC;
    float*    G4   = H3 + (size_t)BB * CC;
    float*    H4   = G4 + (size_t)BB * 256;
    float*    GL   = H4 + (size_t)BB * 256;
    float*    HL   = GL + (size_t)BB * 2;
    // bf16 weight pool (all 16B-aligned: every segment is a multiple of 16B)
    ushort_t* bp    = (ushort_t*)(HL + (size_t)BB * 2);
    ushort_t* gw1b  = bp; bp += (size_t)D2C * CC;
    ushort_t* hw1b  = bp; bp += (size_t)D2C * CC;
    ushort_t* gw3b  = bp; bp += (size_t)CC * CC;
    ushort_t* hw3b  = bp; bp += (size_t)CC * CC;
    ushort_t* gw4b  = bp; bp += (size_t)256 * CC;
    ushort_t* hw4b  = bp; bp += (size_t)256 * CC;
    ushort_t* gwlb  = bp; bp += (size_t)2 * CC;
    ushort_t* hwlb  = bp; bp += (size_t)2 * CC;
    ushort_t* c3wb  = bp; bp += (size_t)CC * D2C;
    ushort_t* c4wb  = bp; bp += (size_t)256 * CC;
    ushort_t* cw1   = bp; bp += (size_t)512 * 1  * D2C;
    ushort_t* cw2   = bp; bp += (size_t)256 * 3  * D2C;
    ushort_t* cw3   = bp; bp += (size_t)64  * 5  * D2C;
    ushort_t* cw4   = bp; bp += (size_t)64  * 7  * D2C;
    ushort_t* cw5   = bp; bp += (size_t)64  * 9  * D2C;
    ushort_t* cw6   = bp; bp += (size_t)64  * 11 * D2C;

    // ---- Weight prep: bf16 conversion (+ [O][C][K]->[O][K][C] for convs) ----
    auto cvt = [&](const float* s, ushort_t* d, int n) {
        cvt_bf16_kernel<<<(n + 255) / 256, 256, 0, stream>>>(s, d, n);
    };
    cvt(c1_gw, gw1b, D2C * CC);  cvt(c1_hw, hw1b, D2C * CC);
    cvt(c3_gw, gw3b, CC * CC);   cvt(c3_hw, hw3b, CC * CC);
    cvt(c4_gw, gw4b, 256 * CC);  cvt(c4_hw, hw4b, 256 * CC);
    cvt(cl_gw, gwlb, 2 * CC);    cvt(cl_hw, hwlb, 2 * CC);
    cvt(c3_w,  c3wb, CC * D2C);  cvt(c4_w,  c4wb, 256 * CC);
    cvt_tr_conv_kernel<<<1024, 256, 0, stream>>>(w1, cw1, 512, 1);
    cvt_tr_conv_kernel<<<1024, 256, 0, stream>>>(w2, cw2, 256, 3);
    cvt_tr_conv_kernel<<<1024, 256, 0, stream>>>(w3, cw3, 64, 5);
    cvt_tr_conv_kernel<<<1024, 256, 0, stream>>>(w4, cw4, 64, 7);
    cvt_tr_conv_kernel<<<1024, 256, 0, stream>>>(w5, cw5, 64, 9);
    cvt_tr_conv_kernel<<<1024, 256, 0, stream>>>(w6, cw6, 64, 11);

    newctx_kernel<<<BB, 256, 0, stream>>>(beta, context, emb_w, emb_b, demb_w, demb_b, ncbf);

    // ---- 8 gate/bias projections as batched WMMA GEMMs ----
    gates_kernel<<<dim3(D2C / 16, BB / 128), 256, 0, stream>>>(ncbf, gw1b, c1_gb, D2C, 1, G1);
    gates_kernel<<<dim3(D2C / 16, BB / 128), 256, 0, stream>>>(ncbf, hw1b, nullptr, D2C, 0, H1);
    gates_kernel<<<dim3(CC / 16,  BB / 128), 256, 0, stream>>>(ncbf, gw3b, c3_gb, CC, 1, G3);
    gates_kernel<<<dim3(CC / 16,  BB / 128), 256, 0, stream>>>(ncbf, hw3b, nullptr, CC, 0, H3);
    gates_kernel<<<dim3(256 / 16, BB / 128), 256, 0, stream>>>(ncbf, gw4b, c4_gb, 256, 1, G4);
    gates_kernel<<<dim3(256 / 16, BB / 128), 256, 0, stream>>>(ncbf, hw4b, nullptr, 256, 0, H4);
    gates_kernel<<<dim3(1,        BB / 128), 256, 0, stream>>>(ncbf, gwlb, cl_gb, 2, 1, GL);
    gates_kernel<<<dim3(1,        BB / 128), 256, 0, stream>>>(ncbf, hwlb, nullptr, 2, 0, HL);

    // ---- Fused per-batch pipeline (~162 KB dynamic LDS, within 320 KB/WGP) ----
    size_t lds_bytes = (size_t)(NN * HS + 32 * TS + 32 * T3S + 32 * T4S) * sizeof(ushort_t);
    main_kernel<<<BB, 256, lds_bytes, stream>>>(
        x, c1_w, c1_b, G1, H1, c3wb, c3_b, G3, H3, c4wb, c4_b, G4, H4,
        cl_w, cl_b, GL, HL,
        cw1, b1, cw2, b2, cw3, b3, cw4, b4, cw5, b5, cw6, b6, out);
}
